// AttentionHead_850403525308
// MI455X (gfx1250) — matrix operand order repaired and checked by reading, not compiled, not run on hardware
//
#include <hip/hip_runtime.h>

// ---------------------------------------------------------------------------
// AttentionHead for MI455X (gfx1250, wave32, WMMA + Tensor Data Mover).
// Q/K/V projections + flash attention. Matmuls: v_wmma_f32_16x16x32_f16.
// K/V tiles staged into LDS by TDM (tensor_load_to_lds, double buffered,
// s_wait_tensorcnt) with hardware row padding for bank-conflict-free
// ds_load_b128 fragment reads. Fallback: cooperative synchronous staging.
// ---------------------------------------------------------------------------

constexpr int BB  = 4;     // batch
constexpr int FF  = 4096;  // sequence
constexpr int DIN = 1024;  // input dim
constexpr int DH  = 128;   // head dim

// padded LDS row strides (in halves): +8 halves (16B) per row -> lanes of a
// ds_load_b128 hit disjoint 4-bank groups (64 banks x 4B)
constexpr int KT_LD = DH + 8;   // 136 halves = 272B per 32-key-tile K row
constexpr int VT_LD = 32 + 8;   // 40 halves = 80B per V^T row (32 keys)
constexpr int PT_LD = 32 + 8;   // 40 halves per P row

typedef __attribute__((ext_vector_type(16))) _Float16 v16h;
typedef __attribute__((ext_vector_type(8)))  _Float16 v8h;
typedef __attribute__((ext_vector_type(8)))  float    v8f;
typedef __attribute__((ext_vector_type(4)))  unsigned int u32x4;
typedef __attribute__((ext_vector_type(4)))  int i32x4;
typedef __attribute__((ext_vector_type(8)))  int i32x8;

union V16U { v16h v; v8h h[2]; };

// A fragment (16xK, K-step 32), 16-bit layout per ISA 7.12.2
__device__ inline v16h load_fragA(const _Float16* base, int ld, int m, int kg) {
  const _Float16* p = base + (size_t)m * ld;
  V16U u;
  u.h[0] = *(const v8h*)(p + kg * 8);
  u.h[1] = *(const v8h*)(p + 16 + kg * 8);
  return u.v;
}

// B fragment (Kx16, K-step 32); base = B^T row-major (rows=N, cols=K)
__device__ inline v16h load_fragB(const _Float16* base, int ld, int n, int kg) {
  const _Float16* p = base + (size_t)n * ld;
  V16U u;
  u.h[0] = *(const v8h*)(p + kg * 8);
  u.h[1] = *(const v8h*)(p + 16 + kg * 8);
  return u.v;
}

__device__ inline v8f wmma_f16(v16h a, v16h b, v8f c) {
  return __builtin_amdgcn_wmma_f32_16x16x32_f16(false, a, false, b,
                                                (short)0, c, false, false);
}

// ---------------------------------------------------------------------------
// Tensor Data Mover plumbing (CDNA5 ISA-8)
// ---------------------------------------------------------------------------
#if defined(__gfx1250__) && __has_builtin(__builtin_amdgcn_tensor_load_to_lds)
#define USE_TDM 1
#endif

#ifdef USE_TDM
__device__ inline void tdm_load(unsigned lds_addr, unsigned long long gaddr,
                                unsigned d1_0, unsigned d1_2, unsigned d1_3,
                                unsigned d1_4, unsigned d1_5) {
  // D# group 0: count=1 | lds_addr[63:32] | global_addr[120:64] | type=2
  u32x4 g0;
  g0[0] = 1u;                                   // count=1, user descriptor
  g0[1] = lds_addr;                             // LDS byte address
  g0[2] = (unsigned)gaddr;                      // global addr lo
  g0[3] = (unsigned)((gaddr >> 32) & 0x01FFFFFFu) | 0x80000000u; // hi + type=2
  // D# group 1: mask/data_size/pad | tensor dims | tile dims | dim0 stride
  i32x8 g1;
  g1[0] = (int)d1_0;  g1[1] = 0;
  g1[2] = (int)d1_2;  g1[3] = (int)d1_3;
  g1[4] = (int)d1_4;  g1[5] = (int)d1_5;
  g1[6] = 0;          g1[7] = 0;
  i32x4 z4 = {0, 0, 0, 0};
#if __clang_major__ >= 23
  i32x8 z8 = {0, 0, 0, 0, 0, 0, 0, 0};
  __builtin_amdgcn_tensor_load_to_lds(g0, g1, z4, z4, z8, 0);
#else
  __builtin_amdgcn_tensor_load_to_lds(g0, g1, z4, z4, 0);
#endif
}

// group1 dword0: data_size=2B (1<<16) | pad_enable (1<<20) | pad_interval | pad_amount(4dw=3)
//   K tile: 1-D 4096 halves, pad 16B every 64 dwords (one 256B row) -> 272B stride
//   V tile: 2-D 32x128, pad 16B every 16 dwords (one 64B row)       -> 80B stride
constexpr unsigned TDM_K_D0 = (1u << 16) | (1u << 20) | (5u << 22) | (3u << 25);
constexpr unsigned TDM_V_D0 = (1u << 16) | (1u << 20) | (3u << 22) | (3u << 25);

__device__ inline unsigned lds_off(const void* p) {
  // generic LDS address: high dword = shared aperture, low dword = LDS offset
  return (unsigned)(unsigned long long)(uintptr_t)p;
}

__device__ inline void tdm_wait_le2() { __builtin_amdgcn_s_wait_tensorcnt(2); }
__device__ inline void tdm_wait_0()   { __builtin_amdgcn_s_wait_tensorcnt(0); }
#endif  // USE_TDM

// Fallback: cooperative synchronous staging into the same padded LDS layout
__device__ inline void stage_sync(_Float16* dK, _Float16* dV,
                                  const _Float16* sK, const _Float16* sV,
                                  int tid) {
  {   // K tile: 32 rows, first 64 of 128 halves (8 x 16B chunks), 256 chunks total
    int row = tid >> 3, c = tid & 7;
    *(v8h*)(dK + row * KT_LD + c * 8) = *(const v8h*)(sK + (size_t)row * DH + c * 8);
  }
#pragma unroll
  for (int j = 0; j < 2; ++j) {  // V tile: 128 rows x 32 halves (4 x 16B), 512 chunks
    int idx = tid * 2 + j;
    int row = idx >> 2, c = idx & 3;
    *(v8h*)(dV + row * VT_LD + c * 8) = *(const v8h*)(sV + (size_t)row * FF + c * 8);
  }
}

// ---------------------------------------------------------------------------
// Stage 0: dtype conversion / weight transpose
// ---------------------------------------------------------------------------
__global__ void cvt_x_kernel(const float* __restrict__ x,
                             _Float16* __restrict__ xh, int n) {
  int i = blockIdx.x * blockDim.x + threadIdx.x;
  if (i < n) xh[i] = (_Float16)x[i];
}

__global__ void cvt_w_kernel(const float* __restrict__ W,
                             _Float16* __restrict__ Wt) {
  int i = blockIdx.x * blockDim.x + threadIdx.x;
  if (i < DIN * DH) {
    int o = i / DIN, k = i % DIN;
    Wt[i] = (_Float16)W[(size_t)k * DH + o];
  }
}

// ---------------------------------------------------------------------------
// Stage 1: Q/K/V projection. One 16x16 output tile per wave.
// ---------------------------------------------------------------------------
__global__ __launch_bounds__(256)
void proj_qkv_kernel(const _Float16* __restrict__ xh,
                     const _Float16* __restrict__ Wt,   // [3][DH][DIN]
                     const float* __restrict__ bq, const float* __restrict__ bk,
                     const float* __restrict__ bv,
                     _Float16* __restrict__ Qh, _Float16* __restrict__ Kh,
                     _Float16* __restrict__ Vt) {
  const int lane = threadIdx.x & 31;
  const int wave = threadIdx.x >> 5;
  const int tile = blockIdx.x * 8 + wave;

  const int TILES_PER_B = (FF / 16) * 24;
  const int b  = tile / TILES_PER_B;
  const int r0 = tile % TILES_PER_B;
  const int mt = r0 / 24;
  const int ot = r0 % 24;
  const int mat = ot >> 3;                     // 0=Q 1=K 2=V
  const int oc  = (ot & 7) * 16;

  const int m  = lane & 15;
  const int kg = lane >> 4;

  const _Float16* A  = xh + (size_t)(b * FF + mt * 16) * DIN;
  const _Float16* Bw = Wt + ((size_t)mat * DH + oc) * DIN;

  v8f acc = {};
#pragma unroll 4
  for (int k = 0; k < DIN; k += 32) {
    v16h af = load_fragA(A + k, DIN, m, kg);
    v16h bf = load_fragB(Bw + k, DIN, m, kg);
    acc = wmma_f16(af, bf, acc);
  }

  const float* bias = (mat == 0) ? bq : (mat == 1) ? bk : bv;
  const float bcol = bias[oc + m];

  if (mat < 2) {
    _Float16* dst = (mat ? Kh : Qh) + (size_t)(b * FF + mt * 16) * DH;
#pragma unroll
    for (int r = 0; r < 8; ++r) {
      int row = kg * 8 + r;
      dst[(size_t)row * DH + oc + m] = (_Float16)(acc[r] + bcol);
    }
  } else {
    _Float16* dst = Vt + (size_t)b * DH * FF;  // [d][key]
#pragma unroll
    for (int r = 0; r < 8; ++r) {
      int row = kg * 8 + r;
      dst[(size_t)(oc + m) * FF + mt * 16 + row] = (_Float16)(acc[r] + bcol);
    }
  }
}

// ---------------------------------------------------------------------------
// Stage 2: flash attention. 8 waves = 128 queries/block; 32-key blocks staged
// in LDS (TDM double-buffered); 8 QK^T WMMAs + online softmax + 8 PV WMMAs.
// ---------------------------------------------------------------------------
__global__ __launch_bounds__(256)
void attn_fwd_kernel(const _Float16* __restrict__ Qh,
                     const _Float16* __restrict__ Kh,
                     const _Float16* __restrict__ Vt,
                     float* __restrict__ out) {
  __shared__ __attribute__((aligned(16))) _Float16 ldsK[2][32 * KT_LD];  // 2x8704B
  __shared__ __attribute__((aligned(16))) _Float16 ldsV[2][DH * VT_LD];  // 2x10240B
  __shared__ __attribute__((aligned(16))) _Float16 ldsP[8][16 * PT_LD];  // 10240B

  const int tid  = threadIdx.x;
  const int lane = tid & 31;
  const int wave = tid >> 5;
  const int b  = blockIdx.x >> 5;
  const int qb = blockIdx.x & 31;
  const int q0 = qb * 128 + wave * 16;
  const int m  = lane & 15;
  const int kg = lane >> 4;

  const float qs = 0.08838834764831845f * 1.4426950408889634f; // 1/sqrt(128)*log2e

  const _Float16* Qp = Qh + (size_t)(b * FF + q0) * DH;
  v16h qf[4];
#pragma unroll
  for (int kk = 0; kk < 4; ++kk) qf[kk] = load_fragA(Qp + kk * 32, DH, m, kg);

  v8f o[8];
  v8f zero = {};
#pragma unroll
  for (int t = 0; t < 8; ++t) o[t] = zero;
  float mrow[8], lrow[8];
#pragma unroll
  for (int r = 0; r < 8; ++r) { mrow[r] = -1e30f; lrow[r] = 0.0f; }

  const _Float16* Kb = Kh + (size_t)b * FF * DH;
  const _Float16* Vb = Vt + (size_t)b * DH * FF;
  _Float16* myP = &ldsP[wave][0];

  const int NB = FF / 32;

#ifdef USE_TDM
  if (wave == 0) {  // prologue: DMA block 0 into buffer 0 (EXEC-independent)
    tdm_load(lds_off(&ldsK[0][0]), (unsigned long long)(uintptr_t)Kb,
             TDM_K_D0, 0x00010010u, 0x10000000u, 1u, 4096u);          // 1-D 8KB
    tdm_load(lds_off(&ldsV[0][0]), (unsigned long long)(uintptr_t)Vb,
             TDM_V_D0, 0x01000010u, 0x00200000u, 128u, 4096u);        // 2-D 128x32
  }
#endif

  for (int i = 0; i < NB; ++i) {
    const int cur = i & 1;
    const int n0  = i * 32;

#ifdef USE_TDM
    if (i + 1 < NB) {
      if (wave == 0) {  // DMA next block into the other buffer
        const int nn = n0 + 32;
        tdm_load(lds_off(&ldsK[cur ^ 1][0]),
                 (unsigned long long)(uintptr_t)(Kb + (size_t)nn * DH),
                 TDM_K_D0, 0x00010010u, 0x10000000u, 1u, 4096u);
        tdm_load(lds_off(&ldsV[cur ^ 1][0]),
                 (unsigned long long)(uintptr_t)(Vb + nn),
                 TDM_V_D0, 0x01000010u, 0x00200000u, 128u, 4096u);
      }
      tdm_wait_le2();   // current buffer's 2 TDM ops (in-order) are done
    } else {
      tdm_wait_0();
    }
#else
    stage_sync(&ldsK[cur][0], &ldsV[cur][0],
               Kb + (size_t)n0 * DH, Vb + n0, tid);
#endif
    __syncthreads();    // staged tiles visible to all waves

    const _Float16* Kt  = &ldsK[cur][0];
    const _Float16* Vtl = &ldsV[cur][0];

    // ---- S = Q * K^T: two 16-key tiles, fragments from LDS ---------------
    v8f s[2];
#pragma unroll
    for (int t = 0; t < 2; ++t) {
      v8f acc = zero;
#pragma unroll
      for (int kk = 0; kk < 4; ++kk) {
        v16h bf = load_fragB(Kt + (size_t)(t * 16) * KT_LD + kk * 32, KT_LD, m, kg);
        acc = wmma_f16(qf[kk], bf, acc);
      }
      s[t] = acc;
    }

    // ---- online softmax over 32 columns ----------------------------------
#pragma unroll
    for (int r = 0; r < 8; ++r) {
      float s0 = s[0][r] * qs;
      float s1 = s[1][r] * qs;
      float rm = fmaxf(s0, s1);
#pragma unroll
      for (int msk = 1; msk <= 8; msk <<= 1)
        rm = fmaxf(rm, __shfl_xor(rm, msk, 32));
      float mnew = fmaxf(mrow[r], rm);
      float corr = __builtin_exp2f(mrow[r] - mnew);
      float p0 = __builtin_exp2f(s0 - mnew);
      float p1 = __builtin_exp2f(s1 - mnew);
      float ps = p0 + p1;
#pragma unroll
      for (int msk = 1; msk <= 8; msk <<= 1)
        ps += __shfl_xor(ps, msk, 32);
      lrow[r] = lrow[r] * corr + ps;
      mrow[r] = mnew;

      int row = kg * 8 + r;
      myP[row * PT_LD + m]      = (_Float16)p0;
      myP[row * PT_LD + 16 + m] = (_Float16)p1;
#pragma unroll
      for (int t = 0; t < 8; ++t) o[t][r] *= corr;
    }

    asm volatile("s_wait_dscnt 0" ::: "memory");  // P stores -> A-frag reload
    v16h pf = load_fragA(myP, PT_LD, m, kg);

    // ---- O += P * V (V^T tile in LDS: rows=d, cols=32 keys) --------------
#pragma unroll
    for (int t = 0; t < 8; ++t) {
      v16h vf = load_fragB(Vtl + (size_t)(t * 16) * VT_LD, VT_LD, m, kg);
      o[t] = wmma_f16(pf, vf, o[t]);
    }

    __syncthreads();    // all reads done before next DMA/stage overwrites
  }

  // ---- epilogue: O / l, fp32 store ---------------------------------------
  float* Op = out + (size_t)(b * FF + q0) * DH;
#pragma unroll
  for (int r = 0; r < 8; ++r) {
    float inv = 1.0f / lrow[r];
    int row = kg * 8 + r;
#pragma unroll
    for (int t = 0; t < 8; ++t)
      Op[(size_t)row * DH + t * 16 + m] = o[t][r] * inv;
  }
}

// ---------------------------------------------------------------------------
// Launch
// ---------------------------------------------------------------------------
extern "C" void kernel_launch(void* const* d_in, const int* in_sizes, int n_in,
                              void* d_out, int out_size, void* d_ws, size_t ws_size,
                              hipStream_t stream) {
  const float* x  = (const float*)d_in[0];
  const float* Wq = (const float*)d_in[1];
  const float* bq = (const float*)d_in[2];
  const float* Wk = (const float*)d_in[3];
  const float* bk = (const float*)d_in[4];
  const float* Wv = (const float*)d_in[5];
  const float* bv = (const float*)d_in[6];
  float* out = (float*)d_out;

  char* ws = (char*)d_ws;
  _Float16* xh = (_Float16*)ws; ws += (size_t)BB * FF * DIN * 2;  // 33.5 MB
  _Float16* Wt = (_Float16*)ws; ws += (size_t)3 * DH * DIN * 2;   // 0.75 MB
  _Float16* Qh = (_Float16*)ws; ws += (size_t)BB * FF * DH * 2;   // 4 MB
  _Float16* Kh = (_Float16*)ws; ws += (size_t)BB * FF * DH * 2;   // 4 MB
  _Float16* Vt = (_Float16*)ws; ws += (size_t)BB * FF * DH * 2;   // 4 MB

  const int nx = BB * FF * DIN;
  cvt_x_kernel<<<(nx + 255) / 256, 256, 0, stream>>>(x, xh, nx);
  const int nw = DIN * DH;
  cvt_w_kernel<<<(nw + 255) / 256, 256, 0, stream>>>(Wq, Wt + 0 * (size_t)DH * DIN);
  cvt_w_kernel<<<(nw + 255) / 256, 256, 0, stream>>>(Wk, Wt + 1 * (size_t)DH * DIN);
  cvt_w_kernel<<<(nw + 255) / 256, 256, 0, stream>>>(Wv, Wt + 2 * (size_t)DH * DIN);

  proj_qkv_kernel<<<BB * (FF / 16) * 24 / 8, 256, 0, stream>>>(
      xh, Wt, bq, bk, bv, Qh, Kh, Vt);

  attn_fwd_kernel<<<BB * (FF / 128), 256, 0, stream>>>(Qh, Kh, Vt, out);
}
